// GraphTransformerTemporal_14267881357525
// MI455X (gfx1250) — matrix operand order repaired and checked by reading, not compiled, not run on hardware
//
#include <hip/hip_runtime.h>
#include <hip/hip_bf16.h>
#include <cstdint>

typedef __attribute__((ext_vector_type(16))) __bf16 v16bf;
typedef __attribute__((ext_vector_type(4)))  __bf16 v4bf;
typedef __attribute__((ext_vector_type(8)))  float  v8f;

#define N_NODES 10000
#define N_EDGES 80000
#define H_HEADS 8
#define C_DIM   256
#define L_LAYERS 4
#define G_IN    768
#define T_ED    32
#define N_B     48
#define T_FD    14
#define N_CLS   2

#define HC (H_HEADS * C_DIM)   // 2048

__device__ __forceinline__ float gelu_f(float x) {
  return 0.5f * x * (1.0f + erff(x * 0.70710678118654752f));
}

// ---------------- fp32 -> bf16 staging, 4 elems/thread (grid-stride) ----------------
__global__ void k_f32_to_bf16_v4(const float* __restrict__ src, __bf16* __restrict__ dst, long n4) {
  long i = (long)blockIdx.x * blockDim.x + threadIdx.x;
  long stride = (long)gridDim.x * blockDim.x;
  for (; i < n4; i += stride) {
    float4 v = ((const float4*)src)[i];
    v4bf o;
    o[0] = (__bf16)v.x; o[1] = (__bf16)v.y; o[2] = (__bf16)v.z; o[3] = (__bf16)v.w;
    ((v4bf*)dst)[i] = o;
  }
}

// weights: fp32 (K x N) -> bf16 transposed (N x K) so GEMM B-staging is contiguous b128
__global__ void k_w_to_bf16_t(const float* __restrict__ w, __bf16* __restrict__ wt, int K, int Nn) {
  long total = (long)K * Nn;
  long i = (long)blockIdx.x * blockDim.x + threadIdx.x;
  long stride = (long)gridDim.x * blockDim.x;
  for (; i < total; i += stride) {
    int n = (int)(i / K), k = (int)(i % K);
    wt[i] = (__bf16)w[(size_t)k * Nn + n];
  }
}

__global__ void k_copy(const float* __restrict__ src, float* __restrict__ dst, int n) {
  int i = blockIdx.x * blockDim.x + threadIdx.x;
  if (i < n) dst[i] = src[i];
}

// ---------------- WMMA bf16 GEMM:  Y = act(A @ B + bias) ----------------
// A:  M x K bf16 row-major.  Bt: N x K bf16 row-major (pre-transposed weights).
// N % 128 == 0, K % 32 == 0. Block: 256 threads = 8 waves. Tile 128x128, K-step 32.
// Wave w: M-quarter = w>>1 (32 rows = 2 substrips of 16), N-half = w&1 (64 cols).
// Double-buffered async staging: tile i+1 issued via GLOBAL_LOAD_ASYNC_TO_LDS_B128
// while tile i is consumed; s_wait_asynccnt 0x2 retires only the oldest tile's loads.
__global__ __launch_bounds__(256) void k_gemm_bf16_wmma(
    const __bf16* __restrict__ A, const __bf16* __restrict__ Bt,
    const float* __restrict__ bias, float* __restrict__ Y,
    int M, int Nn, int K, int act)
{
  constexpr int TM = 128, TN = 128, TK = 32;
  __shared__ __bf16 As[2][TM * TK];      // [buf][row][k]
  __shared__ __bf16 Bs[2][TN * TK];      // [buf][n][k]

  const int tid   = threadIdx.x;
  const int lane  = tid & 31;
  const int wave  = tid >> 5;
  const int mq    = wave >> 1;        // 0..3
  const int nhalf = wave & 1;         // 0..1
  const int m0 = blockIdx.y * TM;
  const int n0 = blockIdx.x * TN;

  v8f zero = {};
  v8f acc[2][4];
  #pragma unroll
  for (int s = 0; s < 2; ++s)
    #pragma unroll
    for (int i = 0; i < 4; ++i) acc[s][i] = zero;

  // A frag (16-bit 16x32): lanes 0-15: row=lane, elems 0..15 = K0..15
  //                        lanes 16-31: row=lane-16, elems 0..15 = K16..31
  const int aK0 = (lane < 16) ? 0 : 16;
  // B frag (32x16): lanes 0-15: col=lane, K0..15; lanes 16-31: col=lane-16, K16..31
  const int bNl = lane & 15;
  const int bK0 = (lane < 16) ? 0 : 16;

  // staging: thread t copies 8 contiguous bf16 (16B) per matrix: row = t>>1, koff = (t&1)*16
  const int sRow  = tid >> 1;
  const int sKoff = (tid & 1) * 16;
  int gmStage = m0 + sRow; if (gmStage >= M) gmStage = M - 1;   // clamp: rows >= M never stored
  const __bf16* gaBase = A  + (size_t)gmStage     * K + sKoff;
  const __bf16* gbBase = Bt + (size_t)(n0 + sRow) * K + sKoff;
  uint32_t ldsA[2], ldsB[2];
  #pragma unroll
  for (int b = 0; b < 2; ++b) {
    ldsA[b] = (uint32_t)(size_t)(void*)&As[b][sRow * TK + sKoff];
    ldsB[b] = (uint32_t)(size_t)(void*)&Bs[b][sRow * TK + sKoff];
  }

  auto issue = [&](int kt, int b) {
    const __bf16* ga = gaBase + kt;
    const __bf16* gb = gbBase + kt;
    asm volatile("global_load_async_to_lds_b128 %0, %1, off"
                 :: "v"(ldsA[b]), "v"(ga) : "memory");
    asm volatile("global_load_async_to_lds_b128 %0, %1, off"
                 :: "v"(ldsB[b]), "v"(gb) : "memory");
    // warm L2 for the tile after this one (global_prefetch_b8; speculative, OOB-safe)
    __builtin_prefetch(ga + TK, 0, 1);
    __builtin_prefetch(gb + TK, 0, 1);
  };

  const int ntiles = K / TK;
  issue(0, 0);                                   // prologue: tile 0 -> buf 0

  for (int it = 0; it < ntiles; ++it) {
    const int cur = it & 1;
    if (it + 1 < ntiles) {
      issue((it + 1) * TK, cur ^ 1);             // overlap: tile i+1 in flight during compute(i)
      asm volatile("s_wait_asynccnt 0x2" ::: "memory");   // retire tile i (oldest 2 ops)
    } else {
      asm volatile("s_wait_asynccnt 0x0" ::: "memory");
    }
    __syncthreads();                             // all waves' tile-i data visible

    v16bf af[2];
    #pragma unroll
    for (int s = 0; s < 2; ++s) {
      const __bf16* ap = &As[cur][(mq * 32 + s * 16 + (lane & 15)) * TK + aK0];
      #pragma unroll
      for (int i = 0; i < 8; ++i) { af[s][i] = ap[i]; af[s][8 + i] = ap[8 + i]; }
    }
    #pragma unroll
    for (int nb = 0; nb < 4; ++nb) {
      v16bf bf;
      const __bf16* bp = &Bs[cur][(nhalf * 64 + nb * 16 + bNl) * TK + bK0];
      #pragma unroll
      for (int i = 0; i < 16; ++i) bf[i] = bp[i];
      #pragma unroll
      for (int s = 0; s < 2; ++s)
        acc[s][nb] = __builtin_amdgcn_wmma_f32_16x16x32_bf16(
            false, af[s], false, bf, (short)0, acc[s][nb], false, false);
    }
    __syncthreads();                             // readers done before buf reused
  }

  // C/D layout: VGPR r -> M = r (lanes 0-15) / r+8 (lanes 16-31); col = lane&15
  #pragma unroll
  for (int s = 0; s < 2; ++s) {
    const int rbase = m0 + mq * 32 + s * 16 + ((lane >= 16) ? 8 : 0);
    #pragma unroll
    for (int nb = 0; nb < 4; ++nb) {
      int col = n0 + nhalf * 64 + nb * 16 + bNl;
      float bv = bias ? bias[col] : 0.0f;
      #pragma unroll
      for (int i = 0; i < 8; ++i) {
        int gm = rbase + i;
        if (gm < M) {
          float v = acc[s][nb][i] + bv;
          if (act == 1) v = gelu_f(v);
          Y[(size_t)gm * Nn + col] = v;
        }
      }
    }
  }
}

// ---------------- GATv2 edge kernels ----------------
__global__ __launch_bounds__(256) void k_edge_scores(
    const int* __restrict__ ei, const float* __restrict__ xl, const float* __restrict__ xr,
    const float* __restrict__ att_l, unsigned* __restrict__ mbuf, float* __restrict__ ebuf,
    int nE, int nT)
{
  int task = blockIdx.x * 8 + (threadIdx.x >> 5);
  if (task >= nT * H_HEADS) return;
  int lane = threadIdx.x & 31;
  int i = task >> 3;
  int h = task & 7;
  int src = (i < nE) ? ei[i]      : (i - nE);
  int dst = (i < nE) ? ei[nE + i] : (i - nE);
  const float* pl = xl + (size_t)src * HC + h * C_DIM;
  const float* pr = xr + (size_t)dst * HC + h * C_DIM;
  const float* pa = att_l + h * C_DIM;
  float sum = 0.f;
  for (int c = lane; c < C_DIM; c += 32) {
    float z = pl[c] + pr[c];
    z = (z > 0.f) ? z : 0.2f * z;
    sum += pa[c] * z;
  }
  #pragma unroll
  for (int off = 16; off; off >>= 1) sum += __shfl_xor(sum, off, 32);
  if (lane == 0) {
    ebuf[task] = sum;
    unsigned bits = __float_as_uint(sum);
    unsigned enc = (sum >= 0.f) ? (bits ^ 0x80000000u) : ~bits;
    atomicMax(&mbuf[dst * H_HEADS + h], enc);
  }
}

__global__ void k_edge_exp(const int* __restrict__ ei, const unsigned* __restrict__ mbuf,
                           float* __restrict__ ebuf, float* __restrict__ sbuf, int nE, int nT)
{
  int idx = blockIdx.x * blockDim.x + threadIdx.x;
  if (idx >= nT * H_HEADS) return;
  int i = idx >> 3, h = idx & 7;
  int dst = (i < nE) ? ei[nE + i] : (i - nE);
  unsigned enc = mbuf[dst * H_HEADS + h];
  unsigned bits = (enc & 0x80000000u) ? (enc ^ 0x80000000u) : ~enc;
  float m = __uint_as_float(bits);
  float a = expf(ebuf[idx] - m);
  ebuf[idx] = a;
  atomicAdd(&sbuf[dst * H_HEADS + h], a);
}

__global__ __launch_bounds__(256) void k_edge_aggregate(
    const int* __restrict__ ei, const float* __restrict__ xl,
    const float* __restrict__ ebuf, const float* __restrict__ sbuf,
    float* __restrict__ agg, int nE, int nT)
{
  int task = blockIdx.x * 8 + (threadIdx.x >> 5);
  if (task >= nT * H_HEADS) return;
  int lane = threadIdx.x & 31;
  int i = task >> 3, h = task & 7;
  int src = (i < nE) ? ei[i]      : (i - nE);
  int dst = (i < nE) ? ei[nE + i] : (i - nE);
  float w = ebuf[task] / sbuf[dst * H_HEADS + h] * (1.0f / H_HEADS);
  const float* pl = xl + (size_t)src * HC + h * C_DIM;
  float* po = agg + (size_t)dst * C_DIM;
  for (int c = lane; c < C_DIM; c += 32) atomicAdd(&po[c], w * pl[c]);
}

// ---------------- x = LN(x + delta + bias?) over C=256 ----------------
__global__ __launch_bounds__(256) void k_residual_ln(
    float* __restrict__ x, const float* __restrict__ delta, const float* __restrict__ bias)
{
  int n = blockIdx.x, c = threadIdx.x;
  __shared__ float red[256];
  float y = x[(size_t)n * C_DIM + c] + delta[(size_t)n * C_DIM + c] + (bias ? bias[c] : 0.f);
  red[c] = y; __syncthreads();
  for (int s = 128; s; s >>= 1) { if (c < s) red[c] += red[c + s]; __syncthreads(); }
  float mean = red[0] * (1.0f / C_DIM); __syncthreads();
  float d = y - mean;
  red[c] = d * d; __syncthreads();
  for (int s = 128; s; s >>= 1) { if (c < s) red[c] += red[c + s]; __syncthreads(); }
  float var = red[0] * (1.0f / C_DIM);
  x[(size_t)n * C_DIM + c] = d * rsqrtf(var + 1e-5f);
}

__global__ __launch_bounds__(256) void k_col_mean(const float* __restrict__ x, float* __restrict__ cm)
{
  int c = blockIdx.x, t = threadIdx.x;
  float s = 0.f;
  for (int n = t; n < N_NODES; n += 256) s += x[(size_t)n * C_DIM + c];
  __shared__ float red[256];
  red[t] = s; __syncthreads();
  for (int k = 128; k; k >>= 1) { if (t < k) red[t] += red[t + k]; __syncthreads(); }
  if (t == 0) cm[c] = red[0] * (1.0f / N_NODES);
}

__global__ __launch_bounds__(512) void k_vn_update(
    const float* __restrict__ cm, const float* __restrict__ vw1, const float* __restrict__ vb1,
    const float* __restrict__ vw2, const float* __restrict__ vb2, float* __restrict__ vn)
{
  __shared__ float h[2 * C_DIM];
  __shared__ float red[C_DIM];
  int t = threadIdx.x;
  float hv = vb1[t];
  for (int c = 0; c < C_DIM; ++c) hv += cm[c] * vw1[c * (2 * C_DIM) + t];
  h[t] = gelu_f(hv);
  __syncthreads();
  float y = 0.f;
  if (t < C_DIM) {
    float u = vb2[t];
    for (int k = 0; k < 2 * C_DIM; ++k) u += h[k] * vw2[k * C_DIM + t];
    y = vn[t] + u;
    red[t] = y;
  }
  __syncthreads();
  for (int s = 128; s; s >>= 1) { if (t < s) red[t] += red[t + s]; __syncthreads(); }
  float mean = red[0] * (1.0f / C_DIM); __syncthreads();
  if (t < C_DIM) { float d = y - mean; red[t] = d * d; }
  __syncthreads();
  for (int s = 128; s; s >>= 1) { if (t < s) red[t] += red[t + s]; __syncthreads(); }
  if (t < C_DIM) vn[t] = (y - mean) * rsqrtf(red[0] * (1.0f / C_DIM) + 1e-5f);
}

__global__ void k_concat_xc(const float* __restrict__ x, const float* __restrict__ vn,
                            float* __restrict__ xc)
{
  long total = (long)N_NODES * 2 * C_DIM;
  long i = (long)blockIdx.x * blockDim.x + threadIdx.x;
  long stride = (long)gridDim.x * blockDim.x;
  for (; i < total; i += stride) {
    long n = i / (2 * C_DIM); int j = (int)(i % (2 * C_DIM));
    xc[i] = (j < C_DIM) ? x[n * C_DIM + j] : vn[j - C_DIM];
  }
}

// ---------------- fused temporal conv stack (one node per block of 64) ----------------
__global__ __launch_bounds__(64) void k_temporal(
    const float* __restrict__ curves,
    const float* __restrict__ c1w, const float* __restrict__ c1b,
    const float* __restrict__ c2w, const float* __restrict__ c2b,
    const float* __restrict__ c3w, const float* __restrict__ c3b,
    const float* __restrict__ fcw, const float* __restrict__ fcb,
    float* __restrict__ ce)
{
  int n = blockIdx.x, t = threadIdx.x;
  __shared__ float t0[N_B];
  __shared__ float t1[16 * N_B];
  __shared__ float t2[32 * N_B];
  __shared__ float mc[64];
  __shared__ float v32[32];
  if (t < N_B) t0[t] = curves[(size_t)n * N_B + t];
  __syncthreads();
  for (int j = t; j < 16 * N_B; j += 64) {           // conv1 k=5 pad=2
    int o = j / N_B, p = j % N_B;
    float s = c1b[o];
    #pragma unroll
    for (int k = 0; k < 5; ++k) { int q = p + k - 2; if (q >= 0 && q < N_B) s += t0[q] * c1w[o * 5 + k]; }
    t1[j] = gelu_f(s);
  }
  __syncthreads();
  for (int j = t; j < 32 * N_B; j += 64) {           // conv2 k=3 pad=1
    int o = j / N_B, p = j % N_B;
    float s = c2b[o];
    for (int ci = 0; ci < 16; ++ci)
      #pragma unroll
      for (int k = 0; k < 3; ++k) { int q = p + k - 1; if (q >= 0 && q < N_B) s += t1[ci * N_B + q] * c2w[(o * 16 + ci) * 3 + k]; }
    t2[j] = gelu_f(s);
  }
  __syncthreads();
  {                                                  // conv3 k=3 pad=1; thread = out channel
    float acc = 0.f;
    for (int p = 0; p < N_B; ++p) {
      float s = c3b[t];
      for (int ci = 0; ci < 32; ++ci)
        #pragma unroll
        for (int k = 0; k < 3; ++k) { int q = p + k - 1; if (q >= 0 && q < N_B) s += t2[ci * N_B + q] * c3w[(t * 32 + ci) * 3 + k]; }
      acc += gelu_f(s);
    }
    mc[t] = acc * (1.0f / N_B);
  }
  __syncthreads();
  if (t < T_ED) {
    float s = fcb[t];
    for (int c = 0; c < 64; ++c) s += mc[c] * fcw[c * T_ED + t];
    v32[t] = gelu_f(s);
  }
  __syncthreads();
  if (t < T_ED) {                                    // LN over 32 in one wave
    float g = v32[t];
    float sm = g;
    #pragma unroll
    for (int off = 16; off; off >>= 1) sm += __shfl_xor(sm, off, 32);
    float mean = sm * (1.0f / T_ED);
    float d = g - mean, sq = d * d;
    #pragma unroll
    for (int off = 16; off; off >>= 1) sq += __shfl_xor(sq, off, 32);
    ce[(size_t)n * T_ED + t] = d * rsqrtf(sq * (1.0f / T_ED) + 1e-5f);
  }
}

__global__ __launch_bounds__(256) void k_fe(
    const float* __restrict__ tf, const float* __restrict__ tpw, const float* __restrict__ tpb,
    float* __restrict__ fe)
{
  int n = blockIdx.x * 8 + (threadIdx.x >> 5);
  if (n >= N_NODES) return;
  int j = threadIdx.x & 31;
  float s = tpb[j];
  const float* p = tf + (size_t)n * T_FD;
  #pragma unroll
  for (int i = 0; i < T_FD; ++i) s += p[i] * tpw[i * T_ED + j];
  float g = gelu_f(s);
  float sm = g;
  #pragma unroll
  for (int off = 16; off; off >>= 1) sm += __shfl_xor(sm, off, 32);
  float mean = sm * (1.0f / T_ED);
  float d = g - mean, sq = d * d;
  #pragma unroll
  for (int off = 16; off; off >>= 1) sq += __shfl_xor(sq, off, 32);
  fe[(size_t)n * T_ED + j] = d * rsqrtf(sq * (1.0f / T_ED) + 1e-5f);
}

__global__ void k_concat_fused(const float* __restrict__ x, const float* __restrict__ ce,
                               const float* __restrict__ fe, float* __restrict__ fused)
{
  const int F = C_DIM + 2 * T_ED;   // 320
  long total = (long)N_NODES * F;
  long i = (long)blockIdx.x * blockDim.x + threadIdx.x;
  long stride = (long)gridDim.x * blockDim.x;
  for (; i < total; i += stride) {
    long n = i / F; int j = (int)(i % F);
    float v;
    if (j < C_DIM) v = x[n * C_DIM + j];
    else if (j < C_DIM + T_ED) v = ce[n * T_ED + (j - C_DIM)];
    else v = fe[n * T_ED + (j - C_DIM - T_ED)];
    fused[i] = v;
  }
}

__global__ void k_final(const float* __restrict__ o1, const float* __restrict__ ow2,
                        const float* __restrict__ ob2, float* __restrict__ out)
{
  int idx = blockIdx.x * blockDim.x + threadIdx.x;
  if (idx >= N_NODES * N_CLS) return;
  int n = idx >> 1, k = idx & 1;
  float s = ob2[k];
  const float* p = o1 + (size_t)n * C_DIM;
  for (int c = 0; c < C_DIM; ++c) s += p[c] * ow2[c * N_CLS + k];
  out[idx] = s;
}

// ======================================================================
extern "C" void kernel_launch(void* const* d_in, const int* in_sizes, int n_in,
                              void* d_out, int out_size, void* d_ws, size_t ws_size,
                              hipStream_t stream)
{
  const float* x_graph = (const float*)d_in[0];
  const int*   ei      = (const int*)  d_in[1];
  const float* curves  = (const float*)d_in[2];
  const float* tfeat   = (const float*)d_in[3];
  const float* in_w = (const float*)d_in[4];
  const float* in_b = (const float*)d_in[5];
  const float* vn0  = (const float*)d_in[6];
  const float* Wl   = (const float*)d_in[7];
  const float* bl   = (const float*)d_in[8];
  const float* Wr   = (const float*)d_in[9];
  const float* br   = (const float*)d_in[10];
  const float* att  = (const float*)d_in[11];
  const float* gb   = (const float*)d_in[12];
  const float* vw1  = (const float*)d_in[13];
  const float* vb1  = (const float*)d_in[14];
  const float* vw2  = (const float*)d_in[15];
  const float* vb2  = (const float*)d_in[16];
  const float* nw1  = (const float*)d_in[17];
  const float* nb1  = (const float*)d_in[18];
  const float* nw2  = (const float*)d_in[19];
  const float* nb2  = (const float*)d_in[20];
  const float* c1w  = (const float*)d_in[21];
  const float* c1b  = (const float*)d_in[22];
  const float* c2w  = (const float*)d_in[23];
  const float* c2b  = (const float*)d_in[24];
  const float* c3w  = (const float*)d_in[25];
  const float* c3b  = (const float*)d_in[26];
  const float* fcw  = (const float*)d_in[27];
  const float* fcb  = (const float*)d_in[28];
  const float* tpw  = (const float*)d_in[29];
  const float* tpb  = (const float*)d_in[30];
  const float* ow1  = (const float*)d_in[31];
  const float* ob1  = (const float*)d_in[32];
  const float* ow2  = (const float*)d_in[33];
  const float* ob2  = (const float*)d_in[34];

  float* ws = (float*)d_ws;
  const int NT = N_NODES + N_EDGES;
  size_t o = 0;
  float* x    = ws + o; o += (size_t)N_NODES * C_DIM;
  float* x2   = ws + o; o += (size_t)N_NODES * C_DIM;
  float* xl   = ws + o; o += (size_t)N_NODES * HC;
  float* xr   = ws + o; o += (size_t)N_NODES * HC;
  float* ebuf = ws + o; o += (size_t)NT * H_HEADS;
  unsigned* mbuf = (unsigned*)(ws + o); o += (size_t)N_NODES * H_HEADS;
  float* sbuf = ws + o; o += (size_t)N_NODES * H_HEADS;
  float* agg  = ws + o; o += (size_t)N_NODES * C_DIM;
  float* xc   = ws + o; o += (size_t)N_NODES * 2 * C_DIM;
  float* h1   = ws + o; o += (size_t)N_NODES * 2 * C_DIM;
  float* ce   = ws + o; o += (size_t)N_NODES * T_ED;
  float* fe   = ws + o; o += (size_t)N_NODES * T_ED;
  float* fused= ws + o; o += (size_t)N_NODES * (C_DIM + 2 * T_ED);
  float* o1   = ws + o; o += (size_t)N_NODES * C_DIM;
  float* vn   = ws + o; o += C_DIM;
  float* cm   = ws + o; o += C_DIM;
  __bf16* abuf = (__bf16*)(ws + o); o += (size_t)N_NODES * G_IN / 2;        // N*768 halves
  __bf16* wbuf = (__bf16*)(ws + o); o += (size_t)C_DIM * HC / 2;            // 524288 halves

  auto cvt = [&](const float* s, __bf16* d, long n) {
    long n4 = n / 4;
    long b = (n4 + 255) / 256; if (b > 2048) b = 2048; if (b < 1) b = 1;
    k_f32_to_bf16_v4<<<(int)b, 256, 0, stream>>>(s, d, n4);
  };
  auto cvtT = [&](const float* w, __bf16* d, int K, int Nn) {
    long n = (long)K * Nn;
    long b = (n + 255) / 256; if (b > 2048) b = 2048; if (b < 1) b = 1;
    k_w_to_bf16_t<<<(int)b, 256, 0, stream>>>(w, d, K, Nn);
  };
  auto gemm = [&](const float* Af, const float* Wf, const float* bias,
                  float* Y, int M, int Nn, int K, int act) {
    cvt(Af, abuf, (long)M * K);
    cvtT(Wf, wbuf, K, Nn);
    dim3 g(Nn / 128, (M + 127) / 128);
    k_gemm_bf16_wmma<<<g, 256, 0, stream>>>(abuf, wbuf, bias, Y, M, Nn, K, act);
  };

  // vn <- vn0
  k_copy<<<1, 256, 0, stream>>>(vn0, vn, C_DIM);

  // x = x_graph @ in_w + in_b
  gemm(x_graph, in_w, in_b, x, N_NODES, C_DIM, G_IN, 0);

  for (int l = 0; l < L_LAYERS; ++l) {
    const float* Wl_l  = Wl  + (size_t)l * C_DIM * HC;
    const float* bl_l  = bl  + (size_t)l * HC;
    const float* Wr_l  = Wr  + (size_t)l * C_DIM * HC;
    const float* br_l  = br  + (size_t)l * HC;
    const float* att_l = att + (size_t)l * H_HEADS * C_DIM;
    const float* gb_l  = gb  + (size_t)l * C_DIM;
    const float* vw1_l = vw1 + (size_t)l * C_DIM * 2 * C_DIM;
    const float* vb1_l = vb1 + (size_t)l * 2 * C_DIM;
    const float* vw2_l = vw2 + (size_t)l * 2 * C_DIM * C_DIM;
    const float* vb2_l = vb2 + (size_t)l * C_DIM;
    const float* nw1_l = nw1 + (size_t)l * 2 * C_DIM * 2 * C_DIM;
    const float* nb1_l = nb1 + (size_t)l * 2 * C_DIM;
    const float* nw2_l = nw2 + (size_t)l * 2 * C_DIM * C_DIM;
    const float* nb2_l = nb2 + (size_t)l * C_DIM;

    gemm(x, Wl_l, bl_l, xl, N_NODES, HC, C_DIM, 0);
    gemm(x, Wr_l, br_l, xr, N_NODES, HC, C_DIM, 0);

    hipMemsetAsync(mbuf, 0, (size_t)N_NODES * H_HEADS * sizeof(unsigned), stream);
    hipMemsetAsync(sbuf, 0, (size_t)N_NODES * H_HEADS * sizeof(float), stream);
    hipMemsetAsync(agg,  0, (size_t)N_NODES * C_DIM   * sizeof(float), stream);

    k_edge_scores<<<NT, 256, 0, stream>>>(ei, xl, xr, att_l, mbuf, ebuf, N_EDGES, NT);
    k_edge_exp<<<(NT * H_HEADS + 255) / 256, 256, 0, stream>>>(ei, mbuf, ebuf, sbuf, N_EDGES, NT);
    k_edge_aggregate<<<NT, 256, 0, stream>>>(ei, xl, ebuf, sbuf, agg, N_EDGES, NT);
    k_residual_ln<<<N_NODES, 256, 0, stream>>>(x, agg, gb_l);

    k_col_mean<<<C_DIM, 256, 0, stream>>>(x, cm);
    k_vn_update<<<1, 512, 0, stream>>>(cm, vw1_l, vb1_l, vw2_l, vb2_l, vn);

    k_concat_xc<<<2048, 256, 0, stream>>>(x, vn, xc);
    gemm(xc, nw1_l, nb1_l, h1, N_NODES, 2 * C_DIM, 2 * C_DIM, 1);
    gemm(h1, nw2_l, nb2_l, x2, N_NODES, C_DIM, 2 * C_DIM, 0);
    k_residual_ln<<<N_NODES, 256, 0, stream>>>(x, x2, (const float*)nullptr);
  }

  k_temporal<<<N_NODES, 64, 0, stream>>>(curves, c1w, c1b, c2w, c2b, c3w, c3b, fcw, fcb, ce);
  k_fe<<<(N_NODES + 7) / 8, 256, 0, stream>>>(tfeat, tpw, tpb, fe);
  k_concat_fused<<<2048, 256, 0, stream>>>(x, ce, fe, fused);
  gemm(fused, ow1, ob1, o1, N_NODES, C_DIM, C_DIM + 2 * T_ED, 1);
  k_final<<<(N_NODES * N_CLS + 255) / 256, 256, 0, stream>>>(o1, ow2, ob2, (float*)d_out);
}
